// GCN_40982577938953
// MI455X (gfx1250) — matrix operand order repaired and checked by reading, not compiled, not run on hardware
//
#include <hip/hip_runtime.h>
#include <hip/hip_bf16.h>

#define N_NODES 100000
#define N_EDGES 1600000
#define NFEAT   256
#define NHID    64
#define NCLASS  16
#define NTILES  (N_NODES / 16)   // 6250, exact

typedef __attribute__((ext_vector_type(16))) _Float16 v16h;
typedef __attribute__((ext_vector_type(8)))  float    v8f;

union F16x16 { v16h v; _Float16 e[16]; float4 q[2]; };
union F32x8  { v8f  v; float    e[8]; };

// ---------------------------------------------------------------- degrees ---
__global__ void gcn_deg_init(float* __restrict__ deg) {
    int i = blockIdx.x * 256 + threadIdx.x;
    if (i < N_NODES) deg[i] = 1.0f;            // self-loop
}

__global__ void gcn_deg_accum(const long long* __restrict__ dst,
                              float* __restrict__ deg) {
    int e = blockIdx.x * 256 + threadIdx.x;
    if (e < N_EDGES) atomicAdd(&deg[(int)dst[e]], 1.0f);
}

__global__ void gcn_deg_rsqrt(float* __restrict__ deg) {
    int i = blockIdx.x * 256 + threadIdx.x;
    if (i < N_NODES) deg[i] = rsqrtf(deg[i]);  // deg >= 1 always
}

// ------------------------------------------------------------ GEMM1 (WMMA) --
// H[100000x64] = X[100000x256] @ W1[256x64], f16 inputs, f32 accumulate.
__global__ __launch_bounds__(256) void gcn_gemm1_wmma(
    const float* __restrict__ X, const float* __restrict__ W1,
    float* __restrict__ H) {
    __shared__ __align__(32) _Float16 sW[NHID * NFEAT];   // W1^T  [n][k], 32KB
    const int t = threadIdx.x;
    for (int i = t; i < NHID * NFEAT; i += 256) {
        int n = i >> 8, k = i & 255;
        sW[i] = (_Float16)W1[k * NHID + n];
    }
    __syncthreads();

    const int wave = t >> 5, lane = t & 31;
    const int tile = blockIdx.x * 8 + wave;               // wave-uniform
    if (tile >= NTILES) return;
    const int row0 = tile * 16;
    const int m    = lane & 15;
    const int half = lane >> 4;

    F32x8 acc[4];
#pragma unroll
    for (int i = 0; i < 4; ++i)
#pragma unroll
        for (int r = 0; r < 8; ++r) acc[i].e[r] = 0.0f;

    const float* xrow = X + (size_t)(row0 + m) * NFEAT;

#pragma unroll
    for (int k0 = 0; k0 < NFEAT; k0 += 32) {
        const int kb = k0 + half * 16;                    // A-frag K layout
        float4 x0 = *(const float4*)(xrow + kb);
        float4 x1 = *(const float4*)(xrow + kb + 4);
        float4 x2 = *(const float4*)(xrow + kb + 8);
        float4 x3 = *(const float4*)(xrow + kb + 12);
        F16x16 a;
        a.e[0]=(_Float16)x0.x; a.e[1]=(_Float16)x0.y; a.e[2]=(_Float16)x0.z; a.e[3]=(_Float16)x0.w;
        a.e[4]=(_Float16)x1.x; a.e[5]=(_Float16)x1.y; a.e[6]=(_Float16)x1.z; a.e[7]=(_Float16)x1.w;
        a.e[8]=(_Float16)x2.x; a.e[9]=(_Float16)x2.y; a.e[10]=(_Float16)x2.z; a.e[11]=(_Float16)x2.w;
        a.e[12]=(_Float16)x3.x; a.e[13]=(_Float16)x3.y; a.e[14]=(_Float16)x3.z; a.e[15]=(_Float16)x3.w;

        const int kbb = k0 + half * 16;                   // B-frag K layout
#pragma unroll
        for (int nt = 0; nt < 4; ++nt) {
            v16h b = *(const v16h*)(&sW[(nt * 16 + m) * NFEAT + kbb]);
            acc[nt].v = __builtin_amdgcn_wmma_f32_16x16x32_f16(
                false, a.v, false, b, (short)0, acc[nt].v, false, false);
        }
    }
#pragma unroll
    for (int nt = 0; nt < 4; ++nt)
#pragma unroll
        for (int r = 0; r < 8; ++r)
            H[(size_t)(row0 + r + 8 * half) * NHID + nt * 16 + m] = acc[nt].e[r];
}

// -------------------------------------------------------- aggregation ------
// out[n][:] = dinv[n]^2 * h[n][:]   (self-loop term; also initializes buffer)
template <int FEAT>
__global__ void gcn_agg_init(const float* __restrict__ h,
                             const float* __restrict__ dinv,
                             float* __restrict__ out) {
    const int G = FEAT / 4;
    int i = blockIdx.x * 256 + threadIdx.x;               // float4 index
    if (i >= N_NODES * G) return;
    int n = i / G;
    float w = dinv[n]; w = w * w;
    float4 hv = ((const float4*)h)[i];
    ((float4*)out)[i] = make_float4(hv.x * w, hv.y * w, hv.z * w, hv.w * w);
}

// out[d][:] += dinv[s]*dinv[d] * h[s][:]  (FEAT/4 lanes per edge, float4 each)
template <int FEAT>
__global__ void gcn_agg_edges(const float* __restrict__ h,
                              const float* __restrict__ dinv,
                              const long long* __restrict__ src,
                              const long long* __restrict__ dst,
                              float* __restrict__ out) {
    const int G = FEAT / 4;
    int tid = blockIdx.x * 256 + threadIdx.x;
    int e = tid / G, c = tid % G;
    if (e >= N_EDGES) return;
    int s = (int)src[e], d = (int)dst[e];
    float w = dinv[s] * dinv[d];
    float4 hv = ((const float4*)(h + (size_t)s * FEAT))[c];
    float* o = out + (size_t)d * FEAT + c * 4;
    atomicAdd(o + 0, hv.x * w);
    atomicAdd(o + 1, hv.y * w);
    atomicAdd(o + 2, hv.z * w);
    atomicAdd(o + 3, hv.w * w);
}

// ------------------------------------------------ bias + ReLU -> f16 -------
__global__ void gcn_relu_bias_f16(const float* __restrict__ a,
                                  const float* __restrict__ b1,
                                  _Float16* __restrict__ o) {
    int i = blockIdx.x * 256 + threadIdx.x;
    if (i >= N_NODES * NHID) return;
    float v = a[i] + b1[i & (NHID - 1)];
    o[i] = (_Float16)(v > 0.0f ? v : 0.0f);
}

// ------------------------------------------------------------ GEMM2 (WMMA) --
// H3[100000x16] = H2[100000x64](f16) @ W2[64x16]
__global__ __launch_bounds__(256) void gcn_gemm2_wmma(
    const _Float16* __restrict__ H2, const float* __restrict__ W2,
    float* __restrict__ H3) {
    __shared__ __align__(32) _Float16 sW[NCLASS * NHID];  // W2^T [n][k], 2KB
    const int t = threadIdx.x;
    for (int i = t; i < NCLASS * NHID; i += 256) {
        int n = i >> 6, k = i & 63;
        sW[i] = (_Float16)W2[k * NCLASS + n];
    }
    __syncthreads();

    const int wave = t >> 5, lane = t & 31;
    const int tile = blockIdx.x * 8 + wave;
    if (tile >= NTILES) return;
    const int row0 = tile * 16;
    const int m    = lane & 15;
    const int half = lane >> 4;

    F32x8 acc;
#pragma unroll
    for (int r = 0; r < 8; ++r) acc.e[r] = 0.0f;

    const _Float16* hrow = H2 + (size_t)(row0 + m) * NHID;

#pragma unroll
    for (int k0 = 0; k0 < NHID; k0 += 32) {
        const int kb = k0 + half * 16;
        F16x16 a;
        a.q[0] = *(const float4*)(hrow + kb);             // 8 f16, K=kb..kb+7
        a.q[1] = *(const float4*)(hrow + kb + 8);         // 8 f16, K=kb+8..+15
        v16h b = *(const v16h*)(&sW[m * NHID + k0 + half * 16]);
        acc.v = __builtin_amdgcn_wmma_f32_16x16x32_f16(
            false, a.v, false, b, (short)0, acc.v, false, false);
    }
#pragma unroll
    for (int r = 0; r < 8; ++r)
        H3[(size_t)(row0 + r + 8 * half) * NCLASS + m] = acc.e[r];
}

// ---------------------------------------- bias + log_softmax (in-place) ----
__global__ void gcn_bias_logsoftmax(float* __restrict__ a,
                                    const float* __restrict__ b2) {
    int n = blockIdx.x * 256 + threadIdx.x;
    if (n >= N_NODES) return;
    float* row = a + (size_t)n * NCLASS;
    float v[NCLASS];
    float mx = -3.4e38f;
#pragma unroll
    for (int j = 0; j < NCLASS; ++j) {
        v[j] = row[j] + b2[j];
        mx = fmaxf(mx, v[j]);
    }
    float s = 0.0f;
#pragma unroll
    for (int j = 0; j < NCLASS; ++j) s += __expf(v[j] - mx);
    float lse = mx + __logf(s);
#pragma unroll
    for (int j = 0; j < NCLASS; ++j) row[j] = v[j] - lse;
}

// ---------------------------------------------------------------- launch ---
extern "C" void kernel_launch(void* const* d_in, const int* in_sizes, int n_in,
                              void* d_out, int out_size, void* d_ws, size_t ws_size,
                              hipStream_t stream) {
    const float*     X  = (const float*)d_in[0];
    const long long* EI = (const long long*)d_in[1];     // int64 [2, E]
    const float*     W1 = (const float*)d_in[2];
    const float*     b1 = (const float*)d_in[3];
    const float*     W2 = (const float*)d_in[4];
    const float*     b2 = (const float*)d_in[5];
    const long long* src = EI;
    const long long* dst = EI + N_EDGES;

    char* ws = (char*)d_ws;
    float* dinv = (float*)ws;                                  // 0.4 MB
    size_t off  = (((size_t)N_NODES * 4) + 255) & ~(size_t)255;
    float* bufA = (float*)(ws + off);                          // 25.6 MB
    float* bufB = bufA + (size_t)N_NODES * NHID;               // 25.6 MB

    float*     H1    = bufA;                 // gemm1 out
    float*     AGG1  = bufB;                 // layer1 aggregation
    _Float16*  H2f16 = (_Float16*)bufA;      // overlays H1 (H1 dead by then)
    float*     H3    = bufB;                 // overlays AGG1 (dead by then)
    float*     OUT   = (float*)d_out;        // layer2 agg + log_softmax

    // degrees -> dinv
    gcn_deg_init <<<(N_NODES + 255) / 256, 256, 0, stream>>>(dinv);
    gcn_deg_accum<<<(N_EDGES + 255) / 256, 256, 0, stream>>>(dst, dinv);
    gcn_deg_rsqrt<<<(N_NODES + 255) / 256, 256, 0, stream>>>(dinv);

    // layer 1: GEMM (WMMA) -> normalized aggregation -> bias+ReLU (-> f16)
    gcn_gemm1_wmma<<<(NTILES + 7) / 8, 256, 0, stream>>>(X, W1, H1);
    gcn_agg_init<NHID> <<<(N_NODES * (NHID / 4) + 255) / 256, 256, 0, stream>>>(H1, dinv, AGG1);
    gcn_agg_edges<NHID><<<(N_EDGES * (NHID / 4) + 255) / 256, 256, 0, stream>>>(H1, dinv, src, dst, AGG1);
    gcn_relu_bias_f16  <<<(N_NODES * NHID + 255) / 256, 256, 0, stream>>>(AGG1, b1, H2f16);

    // layer 2: GEMM (WMMA) -> normalized aggregation -> bias + log_softmax
    gcn_gemm2_wmma<<<(NTILES + 7) / 8, 256, 0, stream>>>(H2f16, W2, H3);
    gcn_agg_init<NCLASS> <<<(N_NODES * (NCLASS / 4) + 255) / 256, 256, 0, stream>>>(H3, dinv, OUT);
    gcn_agg_edges<NCLASS><<<(N_EDGES * (NCLASS / 4) + 255) / 256, 256, 0, stream>>>(H3, dinv, src, dst, OUT);
    gcn_bias_logsoftmax  <<<(N_NODES + 255) / 256, 256, 0, stream>>>(OUT, b2);
}
